// RowLSTM_38139309588544
// MI455X (gfx1250) — matrix operand order repaired and checked by reading, not compiled, not run on hardware
//
#include <hip/hip_runtime.h>
#include <hip/hip_bf16.h>

#define HIDDEN 40
#define NL 7
#define NB 16
#define HH 64
#define WW 64
#define KS 7
#define PADK 3

typedef __attribute__((ext_vector_type(16))) _Float16 v16h;
typedef __attribute__((ext_vector_type(8)))  _Float16 v8h;
typedef __attribute__((ext_vector_type(4)))  _Float16 v4h;
typedef __attribute__((ext_vector_type(8)))  float    v8f;

// CDNA5 hardware V_TANH_F32 (TRANS32 class, co-executes with VALU).
__device__ __forceinline__ float fast_tanh(float x) {
#if __has_builtin(__builtin_amdgcn_tanhf)
    return __builtin_amdgcn_tanhf(x);
#else
    return tanhf(x);
#endif
}

__device__ __forceinline__ float fast_sigmoid(float x) {
#if __has_builtin(__builtin_amdgcn_tanhf)
    return fmaf(0.5f, __builtin_amdgcn_tanhf(0.5f * x), 0.5f);
#else
    return 1.0f / (1.0f + __expf(-x));
#endif
}

// ---------------------------------------------------------------------------
// Kernel 1: fused 7x7 convs -> i_s [B,160,H,W]
// ---------------------------------------------------------------------------
__global__ __launch_bounds__(256) void conv7_kernel(
    const float* __restrict__ inp, const float* __restrict__ tgt,
    const float* __restrict__ w_is, const float* __restrict__ b_is,
    const float* __restrict__ w_cis, const float* __restrict__ b_cis,
    float* __restrict__ i_s)
{
    __shared__ float s_in[7][70];
    __shared__ float s_tg[7][70];
    __shared__ float s_wc[80 * 49];
    __shared__ float s_wm[80 * 49];

    const int b   = blockIdx.x >> 6;
    const int y   = blockIdx.x & 63;
    const int tid = threadIdx.x;

    for (int i = tid; i < 80 * 49; i += 256) {
        s_wc[i] = w_cis[i];
        const int ky = (i % 49) / 7;
        const int kx = i % 7;
        float wv = w_is[i];
        // mask 'A': zero rows below center; center row: zero kx>=center
        if (ky > PADK || (ky == PADK && kx >= PADK)) wv = 0.0f;
        s_wm[i] = wv;
    }
    for (int i = tid; i < 7 * 70; i += 256) {
        const int r = i / 70, cx = i % 70;
        const int yy = y + r - PADK, xx = cx - PADK;
        float vi = 0.0f, vt = 0.0f;
        if (yy >= 0 && yy < HH && xx >= 0 && xx < WW) {
            vi = inp[b * 4096 + yy * 64 + xx];
            vt = tgt[b * 4096 + yy * 64 + xx];
        }
        s_in[r][cx] = vi;
        s_tg[r][cx] = vt;
    }
    __syncthreads();

    for (int idx = tid; idx < 160 * 64; idx += 256) {
        const int ch = idx >> 6;
        const int w  = idx & 63;
        float acc;
        if (ch < 80) {
            acc = b_cis[ch];
            const float* wp = &s_wc[ch * 49];
            #pragma unroll
            for (int ky = 0; ky < 7; ++ky)
                #pragma unroll
                for (int kx = 0; kx < 7; ++kx)
                    acc = fmaf(wp[ky * 7 + kx], s_in[ky][w + kx], acc);
        } else {
            const int cc = ch - 80;
            acc = b_is[cc];
            const float* wp = &s_wm[cc * 49];
            #pragma unroll
            for (int ky = 0; ky < 7; ++ky)
                #pragma unroll
                for (int kx = 0; kx < 7; ++kx)
                    acc = fmaf(wp[ky * 7 + kx], s_tg[ky][w + kx], acc);
        }
        i_s[(b * 160 + ch) * 4096 + y * 64 + w] = acc;
    }
}

// ---------------------------------------------------------------------------
// Kernel 2: convert w_cell [7,160,40] and w_out [256,40] to f16, K padded->64
// ---------------------------------------------------------------------------
__global__ __launch_bounds__(256) void cvtw_kernel(
    const float* __restrict__ w_cell, const float* __restrict__ w_out,
    _Float16* __restrict__ wc16, _Float16* __restrict__ wo16)
{
    const int i = blockIdx.x * 256 + threadIdx.x;
    if (i < 7 * 160 * 64) {
        const int k = i & 63, row = i >> 6;
        wc16[i] = (k < HIDDEN) ? (_Float16)w_cell[row * HIDDEN + k] : (_Float16)0.0f;
    } else {
        const int j = i - 7 * 160 * 64;
        if (j < 256 * 64) {
            const int k = j & 63, o = j >> 6;
            wo16[j] = (k < HIDDEN) ? (_Float16)w_out[o * HIDDEN + k] : (_Float16)0.0f;
        }
    }
}

// ---------------------------------------------------------------------------
// Kernel 3: row-LSTM recurrence. One block (8 wave32) per batch image.
//   h: f16 in WMMA-B layout s_hB[w*64 + c] (K padded 40->64, pad stays 0)
//   c: f32 in [w][40] layout (channel-contiguous, vector-friendly)
//   comb = W_cell @ h tiled with two INDEPENDENT v_wmma accumulators (no
//   WMMA->WMMA RAW chain, avoids the 4-coexec-NOP hazard penalty).
//   hidden emitted as f16 already in B layout [b][y][w][k] for out_kernel.
// ---------------------------------------------------------------------------
__global__ __launch_bounds__(256) void rowlstm_kernel(
    const float* __restrict__ i_s,
    const float* __restrict__ w_rh, const float* __restrict__ b_rh,
    const float* __restrict__ w_rc, const float* __restrict__ b_rc,
    const _Float16* __restrict__ wc16, const float* __restrict__ b_cell,
    _Float16* __restrict__ hid16)
{
    __shared__ float s_c[64 * HIDDEN];                         // [w][ch] 10240 B
    __shared__ float s_comb[160 * 64];                         // 40960 B (also temp)
    __shared__ alignas(32) _Float16 s_hB[64 * 64];             //  8192 B

    const int b    = blockIdx.x;
    const int tid  = threadIdx.x;
    const int lane = tid & 31;
    const int wave = tid >> 5;
    const int lo   = lane & 15;
    const int hi   = lane >> 4;

    for (int i = tid; i < 64 * 64; i += 256) s_hB[i] = (_Float16)0.0f;
    for (int i = tid; i < 64 * HIDDEN; i += 256) s_c[i] = 0.0f;
    __syncthreads();

    for (int y = 0; y < HH; ++y) {
        // prefetch next row's gate inputs (global_prefetch_b8)
        if (y + 1 < HH) {
            for (int ch = tid; ch < 160; ch += 256) {
                __builtin_prefetch(&i_s[(b * 160 + ch) * 4096 + (y + 1) * 64], 0, 1);
            }
        }

        // ---- h = conv1x3(h) + b_rh; temp transposed [w][64] so the f32->f16
        //      copy into s_hB is fully contiguous/vectorized ----
        for (int i = tid; i < 64 * 64; i += 256) {
            const int o = i >> 6, w = i & 63;
            float acc = 0.0f;
            if (o < HIDDEN) {
                acc = b_rh[o];
                const float4* wp4 = (const float4*)&w_rh[o * 120];
                const v4h zz = {};
                #pragma unroll
                for (int cj = 0; cj < 10; ++cj) {
                    const float4 q0 = wp4[cj * 3 + 0];
                    const float4 q1 = wp4[cj * 3 + 1];
                    const float4 q2 = wp4[cj * 3 + 2];
                    const v4h m4 = (w > 0)  ? *(const v4h*)&s_hB[(w - 1) * 64 + cj * 4] : zz;
                    const v4h c4 =            *(const v4h*)&s_hB[w * 64 + cj * 4];
                    const v4h p4 = (w < 63) ? *(const v4h*)&s_hB[(w + 1) * 64 + cj * 4] : zz;
                    acc = fmaf(q0.x, (float)m4[0], fmaf(q0.y, (float)c4[0], fmaf(q0.z, (float)p4[0], acc)));
                    acc = fmaf(q0.w, (float)m4[1], fmaf(q1.x, (float)c4[1], fmaf(q1.y, (float)p4[1], acc)));
                    acc = fmaf(q1.z, (float)m4[2], fmaf(q1.w, (float)c4[2], fmaf(q2.x, (float)p4[2], acc)));
                    acc = fmaf(q2.y, (float)m4[3], fmaf(q2.z, (float)c4[3], fmaf(q2.w, (float)p4[3], acc)));
                }
            }
            s_comb[w * 64 + o] = acc;      // transposed temp (pad lanes write 0)
        }
        __syncthreads();
        for (int j = tid; j < 512; j += 256) {   // 4096 halfs, 8 per thread-iter
            const float4 a = ((const float4*)s_comb)[2 * j + 0];
            const float4 bq = ((const float4*)s_comb)[2 * j + 1];
            v8h h8;
            h8[0] = (_Float16)a.x;  h8[1] = (_Float16)a.y;
            h8[2] = (_Float16)a.z;  h8[3] = (_Float16)a.w;
            h8[4] = (_Float16)bq.x; h8[5] = (_Float16)bq.y;
            h8[6] = (_Float16)bq.z; h8[7] = (_Float16)bq.w;
            *(v8h*)&s_hB[j * 8] = h8;
        }
        __syncthreads();

        // ---- c = conv1x3(c) + b_rc  (c in [w][40] layout) ----
        for (int i = tid; i < HIDDEN * 64; i += 256) {
            const int o = i >> 6, w = i & 63;
            float acc = b_rc[o];
            const float4* wp4 = (const float4*)&w_rc[o * 120];
            const float4 z4 = {0.0f, 0.0f, 0.0f, 0.0f};
            #pragma unroll
            for (int cj = 0; cj < 10; ++cj) {
                const float4 q0 = wp4[cj * 3 + 0];
                const float4 q1 = wp4[cj * 3 + 1];
                const float4 q2 = wp4[cj * 3 + 2];
                const float4 m4 = (w > 0)  ? *(const float4*)&s_c[(w - 1) * 40 + cj * 4] : z4;
                const float4 c4 =            *(const float4*)&s_c[w * 40 + cj * 4];
                const float4 p4 = (w < 63) ? *(const float4*)&s_c[(w + 1) * 40 + cj * 4] : z4;
                acc = fmaf(q0.x, m4.x, fmaf(q0.y, c4.x, fmaf(q0.z, p4.x, acc)));
                acc = fmaf(q0.w, m4.y, fmaf(q1.x, c4.y, fmaf(q1.y, p4.y, acc)));
                acc = fmaf(q1.z, m4.z, fmaf(q1.w, c4.z, fmaf(q2.x, p4.z, acc)));
                acc = fmaf(q2.y, m4.w, fmaf(q2.z, c4.w, fmaf(q2.w, p4.w, acc)));
            }
            s_comb[w * 40 + o] = acc;     // temp, same [w][40] layout
        }
        __syncthreads();
        for (int j = tid; j < 640; j += 256)
            ((float4*)s_c)[j] = ((const float4*)s_comb)[j];
        __syncthreads();

        // ---- 7 stacked LSTM cell layers ----
        for (int l = 0; l < NL; ++l) {
            const _Float16* WA = wc16 + l * 160 * 64;
            // 40 tiles (10 M x 4 N) over 8 waves -> 5 tiles/wave
            for (int t = wave; t < 40; t += 8) {
                const int tm = t >> 2;
                const int tn = t & 3;
                const _Float16* ap = WA + (tm * 16 + lo) * 64 + hi * 16;
                const v8h a00 = *(const v8h*)(ap);
                const v8h a01 = *(const v8h*)(ap + 8);
                const v8h a10 = *(const v8h*)(ap + 32);
                const v8h a11 = *(const v8h*)(ap + 40);
                const v16h av0 = __builtin_shufflevector(a00, a01,
                    0, 1, 2, 3, 4, 5, 6, 7, 8, 9, 10, 11, 12, 13, 14, 15);
                const v16h av1 = __builtin_shufflevector(a10, a11,
                    0, 1, 2, 3, 4, 5, 6, 7, 8, 9, 10, 11, 12, 13, 14, 15);
                const v16h bv0 = *(const v16h*)&s_hB[(tn * 16 + lo) * 64 + hi * 16];
                const v16h bv1 = *(const v16h*)&s_hB[(tn * 16 + lo) * 64 + 32 + hi * 16];
                v8f acc0 = {};
                v8f acc1 = {};
                // independent accumulators: no WMMA->WMMA RAW hazard
                acc0 = __builtin_amdgcn_wmma_f32_16x16x32_f16(
                    false, av0, false, bv0, (short)0, acc0, false, false);
                acc1 = __builtin_amdgcn_wmma_f32_16x16x32_f16(
                    false, av1, false, bv1, (short)0, acc1, false, false);
                #pragma unroll
                for (int r = 0; r < 8; ++r) {
                    const int M = tm * 16 + hi * 8 + r;
                    const int N = tn * 16 + lo;
                    s_comb[M * 64 + N] = acc0[r] + acc1[r];
                }
            }
            __syncthreads();
            // gates, 4 consecutive w per thread (float4 / b128 grain)
            const float* bc = b_cell + l * 160;
            const int base = b * 160 * 4096 + y * 64;
            for (int g = tid; g < 640; g += 256) {
                const int cg = g >> 4;
                const int w4 = (g & 15) << 2;
                const float4 x_i = *(const float4*)&i_s[base + (cg)       * 4096 + w4];
                const float4 x_f = *(const float4*)&i_s[base + (40 + cg)  * 4096 + w4];
                const float4 x_o = *(const float4*)&i_s[base + (80 + cg)  * 4096 + w4];
                const float4 x_g = *(const float4*)&i_s[base + (120 + cg) * 4096 + w4];
                const float4 m_i = *(const float4*)&s_comb[(cg)       * 64 + w4];
                const float4 m_f = *(const float4*)&s_comb[(40 + cg)  * 64 + w4];
                const float4 m_o = *(const float4*)&s_comb[(80 + cg)  * 64 + w4];
                const float4 m_g = *(const float4*)&s_comb[(120 + cg) * 64 + w4];
                const float bi = bc[cg], bf = bc[40 + cg];
                const float bo = bc[80 + cg], bg = bc[120 + cg];
                #pragma unroll
                for (int j = 0; j < 4; ++j) {
                    const int w = w4 + j;
                    const float ig = fast_sigmoid((&x_i.x)[j] + (&m_i.x)[j] + bi);
                    const float fg = fast_sigmoid((&x_f.x)[j] + (&m_f.x)[j] + bf);
                    const float og = fast_sigmoid((&x_o.x)[j] + (&m_o.x)[j] + bo);
                    const float gg = fast_tanh((&x_g.x)[j] + (&m_g.x)[j] + bg);
                    const float cv = fg * s_c[w * 40 + cg] + ig * gg;
                    s_c[w * 40 + cg] = cv;
                    s_hB[w * 64 + cg] = (_Float16)(og * fast_tanh(cv));
                }
            }
            __syncthreads();
        }

        // emit this row's h (already f16, already in B layout) -> hid16
        {
            const uint4* src = (const uint4*)s_hB;                 // 512 x 16B
            uint4* dst = (uint4*)&hid16[(b * 64 + y) * 4096];
            for (int j = tid; j < 512; j += 256) dst[j] = src[j];
        }
        __syncthreads();
    }
}

// ---------------------------------------------------------------------------
// Kernel 4: out = relu(w_out @ hidden + b_out)
//   B fragments read DIRECTLY from global (hid16 already in B layout),
//   no LDS staging, no barrier. One block per (b,y).
// ---------------------------------------------------------------------------
__global__ __launch_bounds__(256) void out_kernel(
    const _Float16* __restrict__ hid16, const _Float16* __restrict__ wo16,
    const float* __restrict__ b_out, float* __restrict__ out)
{
    const int b    = blockIdx.x >> 6;
    const int y    = blockIdx.x & 63;
    const int tid  = threadIdx.x;
    const int lane = tid & 31;
    const int wave = tid >> 5;
    const int lo   = lane & 15;
    const int hi   = lane >> 4;

    const _Float16* hb = hid16 + (b * 64 + y) * 4096;

    // 64 tiles (16 M x 4 N) over 8 waves -> 8 tiles/wave
    for (int t = wave; t < 64; t += 8) {
        const int tm = t >> 2;
        const int tn = t & 3;
        const _Float16* ap = wo16 + (tm * 16 + lo) * 64 + hi * 16;
        const v8h a00 = *(const v8h*)(ap);
        const v8h a01 = *(const v8h*)(ap + 8);
        const v8h a10 = *(const v8h*)(ap + 32);
        const v8h a11 = *(const v8h*)(ap + 40);
        const v16h av0 = __builtin_shufflevector(a00, a01,
            0, 1, 2, 3, 4, 5, 6, 7, 8, 9, 10, 11, 12, 13, 14, 15);
        const v16h av1 = __builtin_shufflevector(a10, a11,
            0, 1, 2, 3, 4, 5, 6, 7, 8, 9, 10, 11, 12, 13, 14, 15);
        const v16h bv0 = *(const v16h*)&hb[(tn * 16 + lo) * 64 + hi * 16];
        const v16h bv1 = *(const v16h*)&hb[(tn * 16 + lo) * 64 + 32 + hi * 16];
        v8f acc0 = {};
        v8f acc1 = {};
        acc0 = __builtin_amdgcn_wmma_f32_16x16x32_f16(
            false, av0, false, bv0, (short)0, acc0, false, false);
        acc1 = __builtin_amdgcn_wmma_f32_16x16x32_f16(
            false, av1, false, bv1, (short)0, acc1, false, false);
        #pragma unroll
        for (int r = 0; r < 8; ++r) {
            const int M = tm * 16 + hi * 8 + r;
            const int N = tn * 16 + lo;
            const float v = acc0[r] + acc1[r] + b_out[M];
            out[(b * 256 + M) * 4096 + y * 64 + N] = v > 0.0f ? v : 0.0f;
        }
    }
}

// ---------------------------------------------------------------------------
extern "C" void kernel_launch(void* const* d_in, const int* in_sizes, int n_in,
                              void* d_out, int out_size, void* d_ws, size_t ws_size,
                              hipStream_t stream)
{
    const float* input  = (const float*)d_in[0];
    const float* target = (const float*)d_in[1];
    const float* w_is   = (const float*)d_in[2];
    const float* b_is   = (const float*)d_in[3];
    const float* w_cis  = (const float*)d_in[4];
    const float* b_cis  = (const float*)d_in[5];
    const float* w_rh   = (const float*)d_in[6];
    const float* b_rh   = (const float*)d_in[7];
    const float* w_rc   = (const float*)d_in[8];
    const float* b_rc   = (const float*)d_in[9];
    const float* w_cell = (const float*)d_in[10];
    const float* b_cell = (const float*)d_in[11];
    const float* w_out  = (const float*)d_in[12];
    const float* b_out  = (const float*)d_in[13];

    char* ws = (char*)d_ws;
    float*     i_s   = (float*)(ws);                    // 16*160*64*64 f32 = 41,943,040 B
    _Float16*  hid16 = (_Float16*)(ws + 41943040);      // 16*64*64*64  f16 =  8,388,608 B
    _Float16*  wc16  = (_Float16*)(ws + 50331648);      // 7*160*64 f16 = 143,360 B
    _Float16*  wo16  = (_Float16*)(ws + 50475008);      // 256*64   f16 =  32,768 B

    conv7_kernel<<<NB * HH, 256, 0, stream>>>(input, target, w_is, b_is,
                                              w_cis, b_cis, i_s);
    cvtw_kernel<<<(7 * 160 * 64 + 256 * 64) / 256, 256, 0, stream>>>(
        w_cell, w_out, wc16, wo16);
    rowlstm_kernel<<<NB, 256, 0, stream>>>(i_s, w_rh, b_rh, w_rc, b_rc,
                                           wc16, b_cell, hid16);
    out_kernel<<<NB * HH, 256, 0, stream>>>(hid16, wo16, b_out, (float*)d_out);
}
